// VectorQuantizer_59227599012564
// MI455X (gfx1250) — matrix operand order repaired and checked by reading, not compiled, not run on hardware
//
#include <hip/hip_runtime.h>
#include <hip/hip_bf16.h>

// ---------------------------------------------------------------------------
// VQ-VAE vector quantizer for gfx1250 (MI455X).
//   inputs   : [32768, 512] f32
//   codebook : [8192, 512] f32
// out = concat( quantized_st [32768*512] f32, loss [1] f32, indices [32768] f32 )
//
// Split-precision bf16 GEMM (hi + lo residual) on V_WMMA_F32_16X16X32_BF16:
//   dot_f32(x, c) ~= xhi*chi + xhi*clo + xlo*chi   (3 independent WMMA chains)
// B tiles are staged global->LDS with the gfx1250 async copy engine
// (GLOBAL_LOAD_ASYNC_TO_LDS_B128, ASYNCcnt), double-buffered so a full
// 16x16x512 tile of prefetch hides L2 latency. A (hi+lo) lives in registers
// (uses the >256-VGPR s_set_vgpr_msb window).
// ---------------------------------------------------------------------------

#define VQ_N     32768
#define VQ_K     8192
#define VQ_D     512
#define KSTEPS   (VQ_D / 32)      // 16 wmma K-steps per 16x16 tile
#define NTILES   (VQ_K / 16)      // 512 codebook column tiles

typedef __attribute__((ext_vector_type(16))) __bf16 v16bf;
typedef __attribute__((ext_vector_type(8)))  __bf16 v8bf;
typedef __attribute__((ext_vector_type(8)))  float  v8f;

union V16U { v16bf v; v8bf h[2]; };

__device__ __forceinline__ unsigned short f32_to_bf16_rne(float f) {
  unsigned int u = __float_as_uint(f);
  unsigned int r = (u + 0x7FFFu + ((u >> 16) & 1u)) >> 16;
  return (unsigned short)r;
}
__device__ __forceinline__ float bf16_bits_to_f32(unsigned short h) {
  return __uint_as_float(((unsigned int)h) << 16);
}

// Async global -> LDS 16-byte transfer (gfx1250, tracked with ASYNCcnt).
__device__ __forceinline__ void async_g2l_b128(unsigned lds_off, const void* gaddr) {
  asm volatile("global_load_async_to_lds_b128 %0, %1, off"
               :: "v"(lds_off), "v"(gaddr) : "memory");
}
__device__ __forceinline__ void wait_async0() {
#if __has_builtin(__builtin_amdgcn_s_wait_asynccnt)
  __builtin_amdgcn_s_wait_asynccnt(0);
#else
  asm volatile("s_wait_asynccnt 0x0" ::: "memory");
#endif
}

// ---------------------------------------------------------------------------
// Kernel 1: per-row prep. Convert f32 rows to bf16 hi + bf16 lo residual,
// and compute the fp32 row norm ||r||^2. One wave (32 lanes) per row.
// ---------------------------------------------------------------------------
__global__ void __launch_bounds__(256)
vq_prep_rows(const float* __restrict__ src, int rows,
             unsigned short* __restrict__ hi, unsigned short* __restrict__ lo,
             float* __restrict__ norm2) {
  int row  = blockIdx.x * 8 + (threadIdx.x >> 5);
  int lane = threadIdx.x & 31;
  if (row >= rows) return;
  const float* r = src + (size_t)row * VQ_D;
  float s = 0.0f;
  size_t base = (size_t)row * VQ_D;
  for (int i = lane; i < VQ_D; i += 32) {
    float x = r[i];
    s += x * x;
    unsigned short h = f32_to_bf16_rne(x);
    float xh = bf16_bits_to_f32(h);
    unsigned short l = f32_to_bf16_rne(x - xh);
    hi[base + i] = h;
    lo[base + i] = l;
  }
  for (int off = 16; off; off >>= 1) s += __shfl_xor(s, off, 32);
  if (lane == 0) norm2[row] = s;
}

// ---------------------------------------------------------------------------
// Kernel 2: distance + argmin. 4 waves/block, 16 rows/wave (64 rows/block).
// A hi+lo in registers; B tiles (16 cols x 512 K, hi+lo = 32 KB) staged into
// LDS by the async copy engine, double buffered (64 KB LDS total).
// ---------------------------------------------------------------------------
__global__ void __launch_bounds__(128)
vq_argmin(const unsigned short* __restrict__ inHi,
          const unsigned short* __restrict__ inLo,
          const unsigned short* __restrict__ cbHi,
          const unsigned short* __restrict__ cbLo,
          const float* __restrict__ x2,
          const float* __restrict__ c2,
          int* __restrict__ amin) {
  // [buffer][plane hi/lo][16 rows * 512 K]  = 2*2*8192*2B = 64 KB
  __shared__ unsigned short bbuf[2][2][16 * VQ_D];

  const int tid  = threadIdx.x;
  const int wave = tid >> 5;
  const int lane = tid & 31;
  const int half = lane >> 4;          // 0: lanes 0-15, 1: lanes 16-31
  const int l15  = lane & 15;
  const int m0   = blockIdx.x * 64 + wave * 16;

  // ---- stage helper: copy one 16x512 hi+lo tile (2 x 16 KB, contiguous) ----
  // 8192 ushorts per plane; 128 threads * 8 iters * 8 ushorts (b128 each).
  auto stage_tile = [&](int t, int buf) {
    const unsigned short* gh = cbHi + (size_t)t * 16 * VQ_D;
    const unsigned short* gl = cbLo + (size_t)t * 16 * VQ_D;
    unsigned short* sh = &bbuf[buf][0][0];
    unsigned short* sl = &bbuf[buf][1][0];
#pragma unroll
    for (int i = 0; i < 8; ++i) {
      int e = (tid + i * 128) * 8;
      async_g2l_b128((unsigned)(size_t)&sh[e], (const void*)(gh + e));
      async_g2l_b128((unsigned)(size_t)&sl[e], (const void*)(gl + e));
    }
  };

  // ---- A hi+lo tiles into registers (per-lane WMMA A layout) ----
  v16bf ahi[KSTEPS], alo[KSTEPS];
  {
    const unsigned short* arh = inHi + (size_t)(m0 + l15) * VQ_D;
    const unsigned short* arl = inLo + (size_t)(m0 + l15) * VQ_D;
#pragma unroll
    for (int kk = 0; kk < KSTEPS; ++kk) {
      V16U uh, ul;
      uh.h[0] = *(const v8bf*)(arh + kk * 32 + half * 8);
      uh.h[1] = *(const v8bf*)(arh + kk * 32 + 16 + half * 8);
      ul.h[0] = *(const v8bf*)(arl + kk * 32 + half * 8);
      ul.h[1] = *(const v8bf*)(arl + kk * 32 + 16 + half * 8);
      ahi[kk] = uh.v;
      alo[kk] = ul.v;
    }
  }

  float x2v[8];
#pragma unroll
  for (int r = 0; r < 8; ++r) x2v[r] = x2[m0 + half * 8 + r];

  float minv[8];
  int   mini[8];
#pragma unroll
  for (int r = 0; r < 8; ++r) { minv[r] = 3.4e38f; mini[r] = 0; }

  // Prologue: stage tile 0, wait, make visible to block.
  stage_tile(0, 0);
  wait_async0();
  __syncthreads();

  for (int t = 0; t < NTILES; ++t) {
    const int buf = t & 1;
    // Kick off async staging of the next tile into the other buffer.
    if (t + 1 < NTILES) stage_tile(t + 1, buf ^ 1);

    // B rows for this lane (column n = t*16 + l15), hi and lo planes.
    const unsigned short* shrow = &bbuf[buf][0][l15 * VQ_D];
    const unsigned short* slrow = &bbuf[buf][1][l15 * VQ_D];

    v8f acc0 = {}, acc1 = {}, acc2 = {};   // 3 independent WMMA chains
#pragma unroll
    for (int kk = 0; kk < KSTEPS; ++kk) {
      const int kb = kk * 32;
      V16U bh, bl;
      bh.h[0] = *(const v8bf*)(shrow + kb + half * 8);
      bh.h[1] = *(const v8bf*)(shrow + kb + 16 + half * 8);
      bl.h[0] = *(const v8bf*)(slrow + kb + half * 8);
      bl.h[1] = *(const v8bf*)(slrow + kb + 16 + half * 8);
      acc0 = __builtin_amdgcn_wmma_f32_16x16x32_bf16(false, ahi[kk], false, bh.v,
                                                     (short)0, acc0, false, false);
      acc1 = __builtin_amdgcn_wmma_f32_16x16x32_bf16(false, ahi[kk], false, bl.v,
                                                     (short)0, acc1, false, false);
      acc2 = __builtin_amdgcn_wmma_f32_16x16x32_bf16(false, alo[kk], false, bh.v,
                                                     (short)0, acc2, false, false);
    }

    // C layout: elem r -> row m0 + half*8 + r, col t*16 + l15.
    const float c2v  = c2[t * 16 + l15];
    const int   nidx = t * 16 + l15;
#pragma unroll
    for (int r = 0; r < 8; ++r) {
      float dot  = acc0[r] + (acc1[r] + acc2[r]);
      float dist = x2v[r] + c2v - 2.0f * dot;
      if (dist < minv[r]) { minv[r] = dist; mini[r] = nidx; }
    }

    // Next tile's staging must be complete (own ASYNCcnt) and visible to
    // every wave (barrier) before anyone reads it or re-stages this buffer.
    wait_async0();
    __syncthreads();
  }

  // Min+index reduction across the 16-lane group (xor stays inside halves).
#pragma unroll
  for (int off = 1; off < 16; off <<= 1) {
#pragma unroll
    for (int r = 0; r < 8; ++r) {
      float ov = __shfl_xor(minv[r], off, 32);
      int   oi = __shfl_xor(mini[r], off, 32);
      if (ov < minv[r] || (ov == minv[r] && oi < mini[r])) {
        minv[r] = ov; mini[r] = oi;
      }
    }
  }
  if (l15 == 0) {
#pragma unroll
    for (int r = 0; r < 8; ++r) amin[m0 + half * 8 + r] = mini[r];
  }
}

// ---------------------------------------------------------------------------
// Kernel 3: gather fp32 codebook rows, straight-through output x + (q - x),
// per-row squared-error partial, and index (as f32). One block per row.
// ---------------------------------------------------------------------------
__global__ void __launch_bounds__(128)
vq_gather(const float* __restrict__ inputs, const float* __restrict__ codebook,
          const int* __restrict__ amin, float* __restrict__ outq,
          float* __restrict__ outidx, float* __restrict__ partials) {
  const int row = blockIdx.x;
  const int idx = amin[row];
  const float* c = codebook + (size_t)idx * VQ_D;
  const float* x = inputs + (size_t)row * VQ_D;
  float* o = outq + (size_t)row * VQ_D;
  float s = 0.0f;
  for (int i = threadIdx.x; i < VQ_D; i += 128) {
    float xv = x[i];
    float q  = c[i];
    float d  = q - xv;
    o[i] = xv + d;          // matches inputs + (quantized - inputs) bitwise
    s += d * d;
  }
  __shared__ float red[4];
  for (int off = 16; off; off >>= 1) s += __shfl_xor(s, off, 32);
  if ((threadIdx.x & 31) == 0) red[threadIdx.x >> 5] = s;
  __syncthreads();
  if (threadIdx.x == 0) {
    partials[row] = red[0] + red[1] + red[2] + red[3];
    outidx[row] = (float)idx;
  }
}

// ---------------------------------------------------------------------------
// Kernel 4: deterministic final loss reduction over 32768 row partials.
// loss = (1 + 0.25) * mean((q - x)^2)   (e/q latent losses are equal).
// ---------------------------------------------------------------------------
__global__ void __launch_bounds__(256)
vq_loss(const float* __restrict__ partials, int n, float scale,
        float* __restrict__ loss) {
  __shared__ float red[256];
  float s = 0.0f;
  for (int i = threadIdx.x; i < n; i += 256) s += partials[i];
  red[threadIdx.x] = s;
  __syncthreads();
  for (int off = 128; off; off >>= 1) {
    if (threadIdx.x < off) red[threadIdx.x] += red[threadIdx.x + off];
    __syncthreads();
  }
  if (threadIdx.x == 0) *loss = red[0] * scale;
}

// ---------------------------------------------------------------------------
extern "C" void kernel_launch(void* const* d_in, const int* in_sizes, int n_in,
                              void* d_out, int out_size, void* d_ws, size_t ws_size,
                              hipStream_t stream) {
  (void)in_sizes; (void)n_in; (void)out_size; (void)ws_size;
  const float* inputs   = (const float*)d_in[0];
  const float* codebook = (const float*)d_in[1];

  char* ws = (char*)d_ws;
  size_t off = 0;
  unsigned short* inHi = (unsigned short*)(ws + off); off += (size_t)VQ_N * VQ_D * 2;
  unsigned short* inLo = (unsigned short*)(ws + off); off += (size_t)VQ_N * VQ_D * 2;
  unsigned short* cbHi = (unsigned short*)(ws + off); off += (size_t)VQ_K * VQ_D * 2;
  unsigned short* cbLo = (unsigned short*)(ws + off); off += (size_t)VQ_K * VQ_D * 2;
  float* x2       = (float*)(ws + off); off += (size_t)VQ_N * 4;
  float* c2       = (float*)(ws + off); off += (size_t)VQ_K * 4;
  int*   amin     = (int*)(ws + off);   off += (size_t)VQ_N * 4;
  float* partials = (float*)(ws + off); off += (size_t)VQ_N * 4;

  float* outq   = (float*)d_out;
  float* lossp  = outq + (size_t)VQ_N * VQ_D;
  float* outidx = lossp + 1;

  vq_prep_rows<<<VQ_N / 8, 256, 0, stream>>>(inputs, VQ_N, inHi, inLo, x2);
  vq_prep_rows<<<VQ_K / 8, 256, 0, stream>>>(codebook, VQ_K, cbHi, cbLo, c2);
  vq_argmin<<<VQ_N / 64, 128, 0, stream>>>(inHi, inLo, cbHi, cbLo, x2, c2, amin);
  vq_gather<<<VQ_N, 128, 0, stream>>>(inputs, codebook, amin, outq, outidx, partials);
  vq_loss<<<1, 256, 0, stream>>>(partials, VQ_N,
                                 1.25f / ((float)VQ_N * (float)VQ_D), lossp);
}
